// MultiHeadAttention_23648089932026
// MI455X (gfx1250) — matrix operand order repaired and checked by reading, not compiled, not run on hardware
//
#include <hip/hip_runtime.h>

// ---------------------------------------------------------------------------
// MHA forward for MI455X (gfx1250, wave32, WMMA 16x16x32 f16 -> f32 acc).
//   P0: convert q/k/v f32->f16 [M,K]; transpose+convert W* -> f16 [N][K]
//   K1: Q = (q@Wq+bq)*scale -> f16 [B*S,HID]; K likewise; V -> f16 [B,H,D,S]
//   K2: attention per (b,h,qtile64): 132KB LDS score tile, softmax, ctx f16
//   K3: out = ctx@Wo + bo -> f32
// GEMMs: 8 wave32s, 128x128 tile, K-step 32, double-buffered LDS fed by the
// Tensor Data Mover (tensor_load_to_lds + s_wait_tensorcnt), one barrier per
// K-step. Falls back to global->VGPR->LDS staging if the builtin is absent.
// ---------------------------------------------------------------------------

typedef __attribute__((ext_vector_type(16))) _Float16      v16h;
typedef __attribute__((ext_vector_type(8)))  float         v8f;
typedef __attribute__((ext_vector_type(4)))  unsigned int  u32x4;
typedef __attribute__((ext_vector_type(8)))  int           i32x8;
typedef __attribute__((ext_vector_type(4)))  int           i32x4;

#define HID   1024
#define HEADS 16
#define DHEAD 64
#define BATCH 4
#define SEQ   1024
#define MTOT  (BATCH * SEQ)
#define PSTR  1032            // attention LDS score-row stride (halfs)
#define LDT   40              // GEMM LDS tile row stride (halfs): 32 + 8 pad
#define TILEH (128 * LDT)     // halfs per tile buffer

#if __has_builtin(__builtin_amdgcn_tensor_load_to_lds) && \
    __has_builtin(__builtin_amdgcn_s_wait_tensorcnt)
#define HAVE_TDM 1
#else
#define HAVE_TDM 0
#endif

union FragA { v16h v; uint4 q[2]; };
union FragC { v8f  v; float f[8]; };
union H8    { uint4 q; _Float16 h[8]; };
union H4    { uint2 u; _Float16 h[4]; };

static __device__ __forceinline__ v8f wmma_f16(v16h a, v16h b, v8f c) {
    return __builtin_amdgcn_wmma_f32_16x16x32_f16(false, a, false, b,
                                                  (short)0, c, false, false);
}

#if HAVE_TDM
// Issue a TDM load of a 128-row x 32-half tile from a row-major f16 tensor
// (row stride 1024 halfs) into LDS, inserting 8 halfs of pad per 64B line
// so the LDS layout is [128][LDT].
static __device__ __forceinline__ void
tdm_load_tile(unsigned lds_off, const _Float16* gtile, unsigned tensor_rows)
{
    const unsigned long long ga = (unsigned long long)(uintptr_t)gtile;
    u32x4 g0;
    g0[0] = 1u;                                   // count=1, user-mode load
    g0[1] = lds_off;                              // lds_addr (bytes)
    g0[2] = (unsigned)(ga & 0xFFFFFFFFu);         // global_addr[31:0]
    g0[3] = (unsigned)((ga >> 32) & 0x01FFFFFFu)  // global_addr[56:32]
          | (2u << 30);                           // type=2 ("image")
    i32x8 g1;
    g1[0] = (int)((1u << 16)     // data_size: 2 bytes
                | (1u << 20)     // pad_enable
                | (3u << 22)     // pad_interval: 16 DWORDs (one 64B line)
                | (3u << 25));   // pad_amount: 4 DWORDs (8 halfs)
    g1[1] = (int)((HID & 0xFFFFu) << 16);                   // tensor_dim0 lo16
    g1[2] = (int)(((HID >> 16) & 0xFFFFu)                   // tensor_dim0 hi16
                | ((tensor_rows & 0xFFFFu) << 16));         // tensor_dim1 lo16
    g1[3] = (int)(((tensor_rows >> 16) & 0xFFFFu)           // tensor_dim1 hi16
                | (32u << 16));                             // tile_dim0 = 32
    g1[4] = (int)128;                                       // tile_dim1 = 128
    g1[5] = (int)HID;                                       // dim0_stride lo32
    g1[6] = 0;
    g1[7] = 0;
    const i32x4 z4 = {0, 0, 0, 0};
#if defined(__clang_major__) && (__clang_major__ >= 23)
    const i32x8 z8 = {0, 0, 0, 0, 0, 0, 0, 0};
    __builtin_amdgcn_tensor_load_to_lds(g0, g1, z4, z4, z8, 0);
#else
    __builtin_amdgcn_tensor_load_to_lds(g0, g1, z4, z4, 0);
#endif
}
#endif

// ---------------------------------------------------------------------------
// P0a: f32 -> f16, same layout.
// ---------------------------------------------------------------------------
__global__ __launch_bounds__(256)
void cvt_kernel(const float* __restrict__ in, _Float16* __restrict__ out)
{
    const size_t i = ((size_t)blockIdx.x * 256 + threadIdx.x) * 4;
    const float4 f = *(const float4*)(in + i);
    H4 o;
    o.h[0] = (_Float16)f.x; o.h[1] = (_Float16)f.y;
    o.h[2] = (_Float16)f.z; o.h[3] = (_Float16)f.w;
    *(uint2*)(out + i) = o.u;
}

// ---------------------------------------------------------------------------
// P0b: W f32 [K][N] -> Wt f16 [N][K] via 64x64 LDS tile.
// ---------------------------------------------------------------------------
__global__ __launch_bounds__(256)
void wt_kernel(const float* __restrict__ W, _Float16* __restrict__ Wt)
{
    __shared__ __align__(16) _Float16 t[64][72];
    const int tid = threadIdx.x;
    const int k0 = blockIdx.x * 64, n0 = blockIdx.y * 64;
    #pragma unroll
    for (int p = 0; p < 4; ++p) {
        const int kk = p * 16 + (tid >> 4);
        const int nc = (tid & 15) * 4;
        const float4 f = *(const float4*)(W + (size_t)(k0 + kk) * HID + n0 + nc);
        t[kk][nc + 0] = (_Float16)f.x;
        t[kk][nc + 1] = (_Float16)f.y;
        t[kk][nc + 2] = (_Float16)f.z;
        t[kk][nc + 3] = (_Float16)f.w;
    }
    __syncthreads();
    #pragma unroll
    for (int i = 0; i < 2; ++i) {
        const int chunk = tid * 2 + i;
        const int n = chunk >> 3, c = chunk & 7;
        H8 o;
        #pragma unroll
        for (int j = 0; j < 8; ++j) o.h[j] = t[c * 8 + j][n];
        *(uint4*)(Wt + (size_t)(n0 + n) * HID + k0 + c * 8) = o.q;
    }
}

// ---------------------------------------------------------------------------
// GEMM: Out = A[4096,1024](f16) x Wt[1024,1024](f16,[N][K]) + bias.
// MODE: 0 = f16 [M,N] (*scale), 1 = f16 per-head transposed [B,H,D,S],
//       2 = f32 [M,N].
// LDS: smem[0..1]=A bufs, smem[2..3]=B bufs, each 128 x LDT halfs.
// ---------------------------------------------------------------------------
template <int MODE>
__global__ __launch_bounds__(256)
void gemm_kernel(const _Float16* __restrict__ A, const _Float16* __restrict__ Wt,
                 const float* __restrict__ bias, void* __restrict__ Out,
                 float scale)
{
    __shared__ __align__(16) _Float16 smem[4 * TILEH];

    const int tid  = threadIdx.x;
    const int wid  = tid >> 5;
    const int lane = tid & 31;
    const int r    = lane & 15;
    const int hi   = lane >> 4;
    const int rowBase = blockIdx.x * 128;
    const int nBase   = blockIdx.y * 128;

#if HAVE_TDM
    const unsigned lds0 = (unsigned)(uintptr_t)smem;  // LDS byte offset
    if (wid == 0) {
        tdm_load_tile(lds0 + 0 * TILEH * 2, A  + (size_t)rowBase * HID, MTOT);
        tdm_load_tile(lds0 + 2 * TILEH * 2, Wt + (size_t)nBase  * HID, HID);
        __builtin_amdgcn_s_wait_tensorcnt(0);
    }
    __syncthreads();
#else
    const int c0 = tid * 2, c1 = tid * 2 + 1;
    const int ar0 = c0 >> 2, ac0 = (c0 & 3) * 8;
    const int ar1 = c1 >> 2, ac1 = (c1 & 3) * 8;
    uint4 ra0, ra1, rb0, rb1;
    ra0 = *(const uint4*)(A  + (size_t)(rowBase + ar0) * HID + ac0);
    ra1 = *(const uint4*)(A  + (size_t)(rowBase + ar1) * HID + ac1);
    rb0 = *(const uint4*)(Wt + (size_t)(nBase  + ar0) * HID + ac0);
    rb1 = *(const uint4*)(Wt + (size_t)(nBase  + ar1) * HID + ac1);
    *(uint4*)(smem + 0 * TILEH + ar0 * LDT + ac0) = ra0;
    *(uint4*)(smem + 0 * TILEH + ar1 * LDT + ac1) = ra1;
    *(uint4*)(smem + 2 * TILEH + ar0 * LDT + ac0) = rb0;
    *(uint4*)(smem + 2 * TILEH + ar1 * LDT + ac1) = rb1;
    __syncthreads();
#endif

    v8f acc[8] = {};
    for (int kt = 0; kt < 32; ++kt) {
        const int buf = kt & 1;
#if HAVE_TDM
        if (kt < 31 && wid == 0) {           // DMA next tiles during WMMA
            const int k1 = (kt + 1) * 32;
            tdm_load_tile(lds0 + (buf ^ 1) * TILEH * 2,
                          A + (size_t)rowBase * HID + k1, MTOT);
            tdm_load_tile(lds0 + (2 + (buf ^ 1)) * TILEH * 2,
                          Wt + (size_t)nBase * HID + k1, HID);
        }
#else
        if (kt < 31) {
            const int k1 = (kt + 1) * 32;
            ra0 = *(const uint4*)(A  + (size_t)(rowBase + ar0) * HID + k1 + ac0);
            ra1 = *(const uint4*)(A  + (size_t)(rowBase + ar1) * HID + k1 + ac1);
            rb0 = *(const uint4*)(Wt + (size_t)(nBase  + ar0) * HID + k1 + ac0);
            rb1 = *(const uint4*)(Wt + (size_t)(nBase  + ar1) * HID + k1 + ac1);
        }
#endif
        const _Float16* As = smem + buf * TILEH;
        const _Float16* Bs = smem + (2 + buf) * TILEH;
        FragA a;
        a.q[0] = *(const uint4*)(As + (wid * 16 + r) * LDT + 16 * hi);
        a.q[1] = *(const uint4*)(As + (wid * 16 + r) * LDT + 16 * hi + 8);
        #pragma unroll
        for (int t = 0; t < 8; ++t) {
            FragA b;
            b.q[0] = *(const uint4*)(Bs + (t * 16 + r) * LDT + 16 * hi);
            b.q[1] = *(const uint4*)(Bs + (t * 16 + r) * LDT + 16 * hi + 8);
            acc[t] = wmma_f16(a.v, b.v, acc[t]);
        }
#if HAVE_TDM
        if (wid == 0) __builtin_amdgcn_s_wait_tensorcnt(0);
        __asm__ volatile("" ::: "memory");
#else
        if (kt < 31) {
            *(uint4*)(smem + (buf ^ 1) * TILEH + ar0 * LDT + ac0) = ra0;
            *(uint4*)(smem + (buf ^ 1) * TILEH + ar1 * LDT + ac1) = ra1;
            *(uint4*)(smem + (2 + (buf ^ 1)) * TILEH + ar0 * LDT + ac0) = rb0;
            *(uint4*)(smem + (2 + (buf ^ 1)) * TILEH + ar1 * LDT + ac1) = rb1;
        }
#endif
        __syncthreads();                     // single barrier per K-step
    }

    #pragma unroll
    for (int t = 0; t < 8; ++t) {
        const int col = nBase + t * 16 + r;
        const float bv = bias[col];
        FragC c; c.v = acc[t];
        #pragma unroll
        for (int j = 0; j < 8; ++j) {
            const int row = rowBase + wid * 16 + j + 8 * hi;
            const float val = (c.f[j] + bv) * scale;
            if (MODE == 0) {
                ((_Float16*)Out)[(size_t)row * HID + col] = (_Float16)val;
            } else if (MODE == 1) {          // V: [B,H,D,S]
                const int bb = row >> 10, s = row & 1023;
                const int h = col >> 6, d = col & 63;
                ((_Float16*)Out)[(((size_t)bb * HEADS + h) * DHEAD + d) * SEQ + s] =
                    (_Float16)val;
            } else {
                ((float*)Out)[(size_t)row * HID + col] = val;
            }
        }
    }
}

// ---------------------------------------------------------------------------
// Attention: one WG (4 waves) per (b, h, 64-query tile). Full 64x1024 f16
// score tile in 132KB dynamic LDS (2 WGs per 320KB WGP).
// ---------------------------------------------------------------------------
__global__ __launch_bounds__(128)
void attn_kernel(const _Float16* __restrict__ Qh, const _Float16* __restrict__ Kh,
                 const _Float16* __restrict__ Vt, const float* __restrict__ Bias,
                 _Float16* __restrict__ Ctx)
{
    extern __shared__ __align__(16) char smem_raw[];
    _Float16 (*P)[PSTR] = (_Float16 (*)[PSTR])smem_raw;
    float* lrow = (float*)(smem_raw + (size_t)64 * PSTR * sizeof(_Float16));

    const int tid  = threadIdx.x;
    const int w    = tid >> 5;
    const int lane = tid & 31;
    const int r    = lane & 15;
    const int hi   = lane >> 4;
    const int qtile = blockIdx.x;
    const int bh    = blockIdx.y;
    const int b = bh >> 4, h = bh & 15;
    const int q0 = qtile * 64 + w * 16;

    FragA qa[2];
    {
        const _Float16* qp = Qh + ((size_t)(b * SEQ + q0 + r)) * HID + h * DHEAD;
        #pragma unroll
        for (int c = 0; c < 2; ++c) {
            qa[c].q[0] = *(const uint4*)(qp + c * 32 + 16 * hi);
            qa[c].q[1] = *(const uint4*)(qp + c * 32 + 16 * hi + 8);
        }
    }

    const size_t biasBase = (size_t)bh * SEQ * SEQ;

    // ---- Pass 1: scores = bias + Q K^T (scale pre-folded into Q) ----
    for (int kt = 0; kt < 64; ++kt) {
        const int key0 = kt * 16;
        FragC acc;
        #pragma unroll
        for (int j = 0; j < 8; ++j)
            acc.f[j] = Bias[biasBase + (size_t)(q0 + j + 8 * hi) * SEQ + key0 + r];
        FragA kb0, kb1;
        const _Float16* kp = Kh + ((size_t)(b * SEQ + key0 + r)) * HID + h * DHEAD;
        kb0.q[0] = *(const uint4*)(kp + 16 * hi);
        kb0.q[1] = *(const uint4*)(kp + 16 * hi + 8);
        kb1.q[0] = *(const uint4*)(kp + 32 + 16 * hi);
        kb1.q[1] = *(const uint4*)(kp + 32 + 16 * hi + 8);
        acc.v = wmma_f16(qa[0].v, kb0.v, acc.v);
        acc.v = wmma_f16(qa[1].v, kb1.v, acc.v);
        #pragma unroll
        for (int j = 0; j < 8; ++j)
            P[w * 16 + j + 8 * hi][key0 + r] = (_Float16)acc.f[j];
    }
    __syncthreads();

    // ---- Pass 2: row softmax (2 lanes per row; halves merged via shfl) ----
    {
        const int row = w * 16 + r;
        float m = -1e30f;
        for (int kk = hi * 512; kk < hi * 512 + 512; kk += 8) {
            H8 u; u.q = *(const uint4*)&P[row][kk];
            #pragma unroll
            for (int j = 0; j < 8; ++j) m = fmaxf(m, (float)u.h[j]);
        }
        m = fmaxf(m, __shfl_xor(m, 16, 32));
        float s = 0.f;
        for (int kk = hi * 512; kk < hi * 512 + 512; kk += 8) {
            H8 u; u.q = *(const uint4*)&P[row][kk];
            H8 o;
            #pragma unroll
            for (int j = 0; j < 8; ++j) {
                const float e = __expf((float)u.h[j] - m);
                s += e;
                o.h[j] = (_Float16)e;
            }
            *(uint4*)&P[row][kk] = o.q;
        }
        s += __shfl_xor(s, 16, 32);
        if (hi == 0) lrow[row] = s;
    }
    __syncthreads();

    // ---- Pass 3: ctx = P x V, normalized by 1/l on writeout ----
    v8f acc[4] = {};
    for (int kb = 0; kb < 32; ++kb) {
        const int k0 = kb * 32;
        FragA pa;
        pa.q[0] = *(const uint4*)&P[w * 16 + r][k0 + 16 * hi];
        pa.q[1] = *(const uint4*)&P[w * 16 + r][k0 + 16 * hi + 8];
        #pragma unroll
        for (int t = 0; t < 4; ++t) {
            FragA vb;
            const _Float16* vp =
                Vt + ((size_t)bh * DHEAD + t * 16 + r) * SEQ + k0 + 16 * hi;
            vb.q[0] = *(const uint4*)(vp);
            vb.q[1] = *(const uint4*)(vp + 8);
            acc[t] = wmma_f16(pa.v, vb.v, acc[t]);
        }
    }
    #pragma unroll
    for (int t = 0; t < 4; ++t) {
        FragC c; c.v = acc[t];
        #pragma unroll
        for (int j = 0; j < 8; ++j) {
            const int row = q0 + j + 8 * hi;
            const float inv = 1.0f / lrow[w * 16 + j + 8 * hi];
            Ctx[((size_t)(b * SEQ + row)) * HID + h * DHEAD + t * 16 + r] =
                (_Float16)(c.f[j] * inv);
        }
    }
}

// ---------------------------------------------------------------------------
extern "C" void kernel_launch(void* const* d_in, const int* in_sizes, int n_in,
                              void* d_out, int out_size, void* d_ws, size_t ws_size,
                              hipStream_t stream)
{
    (void)in_sizes; (void)n_in; (void)out_size; (void)ws_size;

    const float* q         = (const float*)d_in[0];
    const float* k         = (const float*)d_in[1];
    const float* v         = (const float*)d_in[2];
    const float* attn_bias = (const float*)d_in[3];
    const float* Wq = (const float*)d_in[4];
    const float* Wk = (const float*)d_in[5];
    const float* Wv = (const float*)d_in[6];
    const float* Wo = (const float*)d_in[7];
    const float* bq = (const float*)d_in[8];
    const float* bk = (const float*)d_in[9];
    const float* bv = (const float*)d_in[10];
    const float* bo = (const float*)d_in[11];

    const size_t T = (size_t)MTOT * HID;   // 4M halfs
    const size_t WSZ = (size_t)HID * HID;  // 1M halfs
    _Float16* Qc  = (_Float16*)d_ws;
    _Float16* Kc  = Qc + T;
    _Float16* Vc  = Kc + T;
    _Float16* Qh  = Vc + T;
    _Float16* Kh  = Qh + T;
    _Float16* Vt  = Kh + T;
    _Float16* Wtq = Vt + T;
    _Float16* Wtk = Wtq + WSZ;
    _Float16* Wtv = Wtk + WSZ;
    _Float16* Wto = Wtv + WSZ;
    _Float16* Ctx = Qc;                    // Qc dead after projection GEMMs

    cvt_kernel<<<dim3(T / 1024), 256, 0, stream>>>(q, Qc);
    cvt_kernel<<<dim3(T / 1024), 256, 0, stream>>>(k, Kc);
    cvt_kernel<<<dim3(T / 1024), 256, 0, stream>>>(v, Vc);
    wt_kernel<<<dim3(16, 16), 256, 0, stream>>>(Wq, Wtq);
    wt_kernel<<<dim3(16, 16), 256, 0, stream>>>(Wk, Wtk);
    wt_kernel<<<dim3(16, 16), 256, 0, stream>>>(Wv, Wtv);
    wt_kernel<<<dim3(16, 16), 256, 0, stream>>>(Wo, Wto);

    const dim3 ggrid(32, 8), gblk(256);
    const float scale = 0.125f;            // DHEAD^-0.5
    gemm_kernel<0><<<ggrid, gblk, 0, stream>>>(Qc, Wtq, bq, Qh, scale);
    gemm_kernel<0><<<ggrid, gblk, 0, stream>>>(Kc, Wtk, bk, Kh, 1.0f);
    gemm_kernel<1><<<ggrid, gblk, 0, stream>>>(Vc, Wtv, bv, Vt, 1.0f);

    const size_t smbytes = (size_t)64 * PSTR * sizeof(_Float16) + 64 * sizeof(float);
    (void)hipFuncSetAttribute((const void*)attn_kernel,
                              hipFuncAttributeMaxDynamicSharedMemorySize,
                              (int)smbytes);
    attn_kernel<<<dim3(16, 64), 128, smbytes, stream>>>(Qh, Kh, Vt, attn_bias, Ctx);

    gemm_kernel<2><<<ggrid, gblk, 0, stream>>>(Ctx, Wto, bo, d_out, 1.0f);
}
